// SwinTransformerBlock_76235669504702
// MI455X (gfx1250) — matrix operand order repaired and checked by reading, not compiled, not run on hardware
//
#include <hip/hip_runtime.h>
#include <hip/hip_bf16.h>
#include <math.h>

// ---------------- problem constants ----------------
#define B_      64
#define Hh      56
#define Ww      56
#define Cc      384
#define WIN     7
#define SHIFT_  3
#define HEADS   12
#define HD      32
#define HIDDEN_ 1536
#define NTOK    49                 // tokens per window
#define NWIN    64                 // windows per image
#define TOTWIN  (B_ * NWIN)        // 4096
#define MTOT    (TOTWIN * NTOK)    // 200704  (== B*H*W, multiple of 128)
#define QSCALE  0.17677669529663687f   // 1/sqrt(32)

typedef __attribute__((ext_vector_type(16))) _Float16 v16h;
typedef __attribute__((ext_vector_type(8)))  float    v8f;

union U32H2 { unsigned u; _Float16 h[2]; };
union AF    { v16h v; unsigned u[8]; };

__device__ __forceinline__ v8f wmma_f16(v16h a, v16h b, v8f c) {
  // v_wmma_f32_16x16x32_f16  D = A*B + C, f32 accumulate
  return __builtin_amdgcn_wmma_f32_16x16x32_f16(false, a, false, b, (short)0, c, false, false);
}

__device__ __forceinline__ float wave_sum(float v) {
  #pragma unroll
  for (int o = 16; o; o >>= 1) v += __shfl_xor(v, o, 32);
  return v;
}
__device__ __forceinline__ float wave_max(float v) {
  #pragma unroll
  for (int o = 16; o; o >>= 1) v = fmaxf(v, __shfl_xor(v, o, 32));
  return v;
}

// LDS byte offset (for async-to-LDS VDST operand): generic LDS address is
// {shared_aperture, 32'offset}; low 32 bits are the DS offset.
__device__ __forceinline__ unsigned lds_off(const void* p) {
  return (unsigned)(size_t)p;
}

// async copy of 16 bytes global -> LDS (per lane), ASYNCcnt-tracked
__device__ __forceinline__ void async_b128(unsigned loff, const void* gp) {
  asm volatile("global_load_async_to_lds_b128 %0, %1, off"
               :: "v"(loff), "v"(gp) : "memory");
}

// ---------------- weight cast: fp32 -> f16, layout preserved [Nout,K] ----------------
// (row-major [Nout,K] is already B^T: contiguous K per output column)
__global__ __launch_bounds__(256) void wcvt_kernel(const float* __restrict__ src,
                                                   _Float16* __restrict__ dst,
                                                   int total) {
  int i = blockIdx.x * 256 + threadIdx.x;
  if (i < total) dst[i] = (_Float16)src[i];
}

// ---------------- LN1 + roll(-3,-3) + window partition -> f16 ----------------
__global__ __launch_bounds__(256) void ln1_shift_kernel(const float* __restrict__ x,
                                                        const float* __restrict__ gw,
                                                        const float* __restrict__ bw,
                                                        _Float16* __restrict__ wins) {
  int tid = threadIdx.x, lane = tid & 31, wv = tid >> 5;
  int token = blockIdx.x * 8 + wv;
  if (token >= MTOT) return;
  int win = token / NTOK, n = token - win * NTOK;
  int b = win >> 6, widx = win & 63, wh = widx >> 3, ww = widx & 7;
  int r = n / 7, c = n - r * 7;
  int sr = (wh * 7 + r + SHIFT_) % 56;
  int sc = (ww * 7 + c + SHIFT_) % 56;
  const float* px = x + ((size_t)b * 3136 + sr * 56 + sc) * Cc;
  float v[12], s = 0.f, s2 = 0.f;
  #pragma unroll
  for (int i = 0; i < 12; ++i) { float f = px[lane + 32 * i]; v[i] = f; s += f; s2 += f * f; }
  s = wave_sum(s); s2 = wave_sum(s2);
  float mu = s * (1.f / Cc);
  float var = s2 * (1.f / Cc) - mu * mu;
  float rs = rsqrtf(var + 1e-5f);
  _Float16* po = wins + (size_t)token * Cc;
  #pragma unroll
  for (int i = 0; i < 12; ++i) {
    int ch = lane + 32 * i;
    po[ch] = (_Float16)((v[i] - mu) * rs * gw[ch] + bw[ch]);
  }
}

// ---------------- LN2 (straight layout) -> f16 ----------------
__global__ __launch_bounds__(256) void ln2_kernel(const float* __restrict__ x1,
                                                  const float* __restrict__ gw,
                                                  const float* __restrict__ bw,
                                                  _Float16* __restrict__ h2) {
  int tid = threadIdx.x, lane = tid & 31, wv = tid >> 5;
  int token = blockIdx.x * 8 + wv;
  if (token >= MTOT) return;
  const float* px = x1 + (size_t)token * Cc;
  float v[12], s = 0.f, s2 = 0.f;
  #pragma unroll
  for (int i = 0; i < 12; ++i) { float f = px[lane + 32 * i]; v[i] = f; s += f; s2 += f * f; }
  s = wave_sum(s); s2 = wave_sum(s2);
  float mu = s * (1.f / Cc);
  float var = s2 * (1.f / Cc) - mu * mu;
  float rs = rsqrtf(var + 1e-5f);
  _Float16* po = h2 + (size_t)token * Cc;
  #pragma unroll
  for (int i = 0; i < 12; ++i) {
    int ch = lane + 32 * i;
    po[ch] = (_Float16)((v[i] - mu) * rs * gw[ch] + bw[ch]);
  }
}

// ---------------- 128x64-tile WMMA GEMM, f16 in, f32 accumulate ----------------
// A: [M,K] row-major f16. Bw: [Nout,K] row-major f16 (== B^T).
// Double-buffered async global->LDS staging (ASYNCcnt).
// mode 0: QKV   (o16a/b/c = q,k,v scatter [win,head,tok,hd]; q pre-scaled)
// mode 1: PROJ  (ofp = x1 = x + out, window-reverse + unshift;  res = x)
// mode 2: FC1   (o16a = gelu(out))
// mode 3: FC2   (ofp = out + res;  res = x1)
__global__ __launch_bounds__(256) void gemm128x64_kernel(
    const _Float16* __restrict__ A, const _Float16* __restrict__ Bw,
    const float* __restrict__ bias, int K, int Nn, int mode,
    _Float16* __restrict__ o16a, _Float16* __restrict__ o16b, _Float16* __restrict__ o16c,
    float* __restrict__ ofp, const float* __restrict__ res) {
  __shared__ _Float16 aS[2][128 * 32];   // A tile  [row][k]
  __shared__ _Float16 bS[2][64 * 32];    // B tile  [col][k]  (weights already K-contig per col)

  int tid = threadIdx.x, lane = tid & 31, wv = tid >> 5;
  int g = lane >> 4, l15 = lane & 15;
  int bM = blockIdx.y * 128, bN = blockIdx.x * 64;

  v8f acc[4] = {v8f{}, v8f{}, v8f{}, v8f{}};

  const int arow = tid >> 1, aseg = (tid & 1) * 16;   // 2 threads / A row, 32B each
  const int bcol = tid >> 2, bseg = (tid & 3) * 8;    // 4 threads / B col, 16B each

  const _Float16* gA = A + (size_t)(bM + arow) * K + aseg;
  const _Float16* gB = Bw + (size_t)(bN + bcol) * K + bseg;
  const int steps = K >> 5;

  // issue async copies for K-step s into buffer buf (3 x b128 per thread)
  auto issue = [&](int s, int buf) {
    const _Float16* ga = gA + s * 32;
    async_b128(lds_off(&aS[buf][arow * 32 + aseg]), ga);
    async_b128(lds_off(&aS[buf][arow * 32 + aseg + 8]), ga + 8);
    async_b128(lds_off(&bS[buf][bcol * 32 + bseg]), gB + s * 32);
    if (s + 2 < steps) __builtin_prefetch(ga + 64, 0, 0);  // global_prefetch_b8
  };

  issue(0, 0);
  for (int s = 0; s < steps; ++s) {
    int buf = s & 1;
    if (s + 1 < steps) {
      issue(s + 1, buf ^ 1);
      asm volatile("s_wait_asynccnt 0x3" ::: "memory");   // tile s landed; s+1 in flight
    } else {
      asm volatile("s_wait_asynccnt 0x0" ::: "memory");
    }
    __syncthreads();

    // A fragment: 16x32 f16, contiguous k = 16*(lane/16) + s per lane (B uses the same order)
    AF au;
    int arow2 = wv * 16 + l15;
    #pragma unroll
    for (int i = 0; i < 8; ++i) {
      int e = 2 * i;
      int kk = e + g * 16;
      au.u[i] = *(const unsigned*)&aS[buf][arow2 * 32 + kk];
    }
    #pragma unroll
    for (int t = 0; t < 4; ++t) {
      AF bu;
      int col = t * 16 + l15;
      #pragma unroll
      for (int i = 0; i < 8; ++i) {
        int kk = 2 * i + g * 16;
        bu.u[i] = *(const unsigned*)&bS[buf][col * 32 + kk];
      }
      acc[t] = wmma_f16(au.v, bu.v, acc[t]);
    }
    __syncthreads();   // buffer `buf` free for reuse at step s+2
  }

  // epilogue; D layout: reg r -> row = r + (lane/16)*8, col = lane%16
  #pragma unroll
  for (int t = 0; t < 4; ++t) {
    #pragma unroll
    for (int r = 0; r < 8; ++r) {
      int row = wv * 16 + r + g * 8;
      int col = t * 16 + l15;
      int m = bM + row;
      int j = bN + col;
      float val = acc[t][r] + bias[j];
      if (mode == 0) {             // QKV scatter
        int which = j / (HEADS * HD);
        int head  = (j >> 5) % HEADS;
        int hd    = j & 31;
        int win = m / NTOK, n = m - win * NTOK;
        size_t di = ((size_t)(win * HEADS + head) * NTOK + n) * HD + hd;
        if (which == 0)      o16a[di] = (_Float16)(val * QSCALE);
        else if (which == 1) o16b[di] = (_Float16)val;
        else                 o16c[di] = (_Float16)val;
      } else if (mode == 1) {      // proj + window-reverse + roll(+3,+3) + residual
        int win = m / NTOK, n = m - win * NTOK;
        int b = win >> 6, widx = win & 63, wh = widx >> 3, ww = widx & 7;
        int rr = n / 7, cc = n - rr * 7;
        int fr = (wh * 7 + rr + SHIFT_) % 56;
        int fc = (ww * 7 + cc + SHIFT_) % 56;
        size_t di = ((size_t)b * 3136 + fr * 56 + fc) * Cc + j;
        ofp[di] = res[di] + val;
      } else if (mode == 2) {      // FC1 + exact gelu
        float ge = 0.5f * val * (1.0f + erff(val * 0.70710678118654752f));
        o16a[(size_t)m * HIDDEN_ + j] = (_Float16)ge;
      } else {                     // FC2 + residual
        size_t di = (size_t)m * Cc + j;
        ofp[di] = res[di] + val;
      }
    }
  }
}

// ---------------- fused shifted-window attention: one block per (window, head) ----------------
__global__ __launch_bounds__(256) void attn_kernel(const _Float16* __restrict__ qb,
                                                   const _Float16* __restrict__ kb,
                                                   const _Float16* __restrict__ vb,
                                                   const float* __restrict__ rel,
                                                   _Float16* __restrict__ aout) {
  __shared__ _Float16 qs[64 * 32];   // [tok][hd], rows 49..63 zero
  __shared__ _Float16 ks[64 * 32];
  __shared__ _Float16 vT[32 * 64];   // transposed: [hd][tok]
  __shared__ float    ss[64 * 64];   // attention scores f32
  __shared__ _Float16 ps[64 * 64];   // softmax probs f16

  int tid = threadIdx.x, lane = tid & 31, wv = tid >> 5;
  int g = lane >> 4, l15 = lane & 15;
  int wh_id = blockIdx.x;
  int win = wh_id / HEADS, head = wh_id - win * HEADS;
  size_t base = (size_t)wh_id * NTOK * HD;

  // stage q,k into LDS; v transposed; pad token rows >= 49 with zeros
  for (int u = tid; u < 1024; u += 256) {
    int h0 = u * 2;
    int row = h0 >> 5, coff = h0 & 31;
    unsigned qv = 0u, kv = 0u, vvv = 0u;
    if (row < NTOK) {
      qv  = *(const unsigned*)(qb + base + (size_t)row * HD + coff);
      kv  = *(const unsigned*)(kb + base + (size_t)row * HD + coff);
      vvv = *(const unsigned*)(vb + base + (size_t)row * HD + coff);
    }
    *(unsigned*)&qs[h0] = qv;
    *(unsigned*)&ks[h0] = kv;
    U32H2 uv; uv.u = vvv;
    vT[coff * 64 + row]       = uv.h[0];
    vT[(coff + 1) * 64 + row] = uv.h[1];
  }
  __syncthreads();

  int widx = win & 63, whh = widx >> 3, www = widx & 7;

  // S = (q*scale) @ k^T : 16 tiles of 16x16, K = HD = 32 -> one WMMA each
  for (int rep = 0; rep < 2; ++rep) {
    int tile = wv + rep * 8;
    int tm = tile >> 2, tn = tile & 3;
    AF au, bu;
    int qrow = tm * 16 + l15;
    #pragma unroll
    for (int i = 0; i < 8; ++i) {
      int e = 2 * i;
      int kk = e + g * 16;
      au.u[i] = *(const unsigned*)&qs[qrow * 32 + kk];
    }
    int ktok = tn * 16 + l15;    // B = k^T: fragment col = key token, K dim = hd
    #pragma unroll
    for (int i = 0; i < 8; ++i) {
      int kk = 2 * i + g * 16;
      bu.u[i] = *(const unsigned*)&ks[ktok * 32 + kk];
    }
    v8f c = {};
    c = wmma_f16(au.v, bu.v, c);
    #pragma unroll
    for (int r = 0; r < 8; ++r) {
      int row = tm * 16 + r + g * 8;   // query token
      int col = tn * 16 + l15;         // key token
      float sv;
      if (row < NTOK && col < NTOK) {
        int ri = row / 7, ci = row - ri * 7;
        int rj = col / 7, cj = col - rj * 7;
        int relid = (ri - rj + 6) * 13 + (ci - cj + 6);
        float bias = rel[relid * HEADS + head];
        int hi = whh * 7 + ri, wi = www * 7 + ci;
        int hj = whh * 7 + rj, wj = www * 7 + cj;
        int cnti = (hi < 49 ? 0 : (hi < 53 ? 1 : 2)) * 3 + (wi < 49 ? 0 : (wi < 53 ? 1 : 2));
        int cntj = (hj < 49 ? 0 : (hj < 53 ? 1 : 2)) * 3 + (wj < 49 ? 0 : (wj < 53 ? 1 : 2));
        float mk = (cnti == cntj) ? 0.f : -100.f;
        sv = c[r] + bias + mk;
      } else {
        sv = -1e30f;                   // padding -> softmax weight 0
      }
      ss[row * 64 + col] = sv;
    }
  }
  __syncthreads();

  // row softmax over 64 (padded) keys, wave32 shuffle reductions
  for (int row = wv; row < 64; row += 8) {
    float a0 = ss[row * 64 + lane];
    float a1 = ss[row * 64 + lane + 32];
    float m = wave_max(fmaxf(a0, a1));
    float e0 = __expf(a0 - m), e1 = __expf(a1 - m);
    float s = wave_sum(e0 + e1);
    float inv = 1.0f / s;
    ps[row * 64 + lane]      = (_Float16)(e0 * inv);
    ps[row * 64 + lane + 32] = (_Float16)(e1 * inv);
  }
  __syncthreads();

  // O = P @ V : 8 tiles (4 M x 2 N), K = 64 -> two WMMA steps
  {
    int tm = wv >> 1, tn = wv & 1;
    v8f c = {};
    #pragma unroll
    for (int ks_ = 0; ks_ < 2; ++ks_) {
      int k0 = ks_ * 32;
      AF au, bu;
      int prow = tm * 16 + l15;
      #pragma unroll
      for (int i = 0; i < 8; ++i) {
        int e = 2 * i;
        int kk = k0 + e + g * 16;
        au.u[i] = *(const unsigned*)&ps[prow * 64 + kk];
      }
      int hdcol = tn * 16 + l15;
      #pragma unroll
      for (int i = 0; i < 8; ++i) {
        int kt = k0 + 2 * i + g * 16;
        bu.u[i] = *(const unsigned*)&vT[hdcol * 64 + kt];
      }
      c = wmma_f16(au.v, bu.v, c);
    }
    #pragma unroll
    for (int r = 0; r < 8; ++r) {
      int row = tm * 16 + r + g * 8;
      if (row < NTOK) {
        int col = tn * 16 + l15;
        aout[((size_t)win * NTOK + row) * Cc + head * HD + col] = (_Float16)c[r];
      }
    }
  }
}

// ---------------- host-side orchestration ----------------
extern "C" void kernel_launch(void* const* d_in, const int* in_sizes, int n_in,
                              void* d_out, int out_size, void* d_ws, size_t ws_size,
                              hipStream_t stream) {
  (void)in_sizes; (void)n_in; (void)out_size; (void)ws_size;
  const float* x       = (const float*)d_in[0];
  const float* norm1_g = (const float*)d_in[1];
  const float* norm1_b = (const float*)d_in[2];
  const float* qkv_w   = (const float*)d_in[3];
  const float* qkv_b   = (const float*)d_in[4];
  const float* rel_tab = (const float*)d_in[5];
  const float* proj_w  = (const float*)d_in[6];
  const float* proj_b  = (const float*)d_in[7];
  const float* norm2_g = (const float*)d_in[8];
  const float* norm2_b = (const float*)d_in[9];
  const float* fc1_w   = (const float*)d_in[10];
  const float* fc1_b   = (const float*)d_in[11];
  const float* fc2_w   = (const float*)d_in[12];
  const float* fc2_b   = (const float*)d_in[13];
  float* out = (float*)d_out;

  // workspace carve-out (256B aligned)
  char* wsp = (char*)d_ws;
  auto carve = [&](size_t bytes) -> char* {
    char* p = wsp;
    wsp += (bytes + 255) & ~(size_t)255;
    return p;
  };
  _Float16* wqF  = (_Float16*)carve((size_t)1152 * Cc * 2);                  // [1152,384] f16
  _Float16* wpF  = (_Float16*)carve((size_t)Cc * Cc * 2);                    // [384,384]
  _Float16* wf1F = (_Float16*)carve((size_t)HIDDEN_ * Cc * 2);               // [1536,384]
  _Float16* wf2F = (_Float16*)carve((size_t)Cc * HIDDEN_ * 2);               // [384,1536]
  _Float16* wins = (_Float16*)carve((size_t)MTOT * Cc * 2);                  // also attn_out (aliased)
  _Float16* qbuf = (_Float16*)carve((size_t)TOTWIN * HEADS * NTOK * HD * 2); // also h2 (aliased)
  _Float16* kbuf = (_Float16*)carve((size_t)TOTWIN * HEADS * NTOK * HD * 2);
  _Float16* vbuf = (_Float16*)carve((size_t)TOTWIN * HEADS * NTOK * HD * 2);
  float*    x1   = (float*)   carve((size_t)MTOT * Cc * 4);
  _Float16* fc1o = (_Float16*)carve((size_t)MTOT * HIDDEN_ * 2);
  _Float16* attn_out = wins;   // wins no longer needed once QKV GEMM ran
  _Float16* h2       = qbuf;   // q no longer needed after attention

  // 1) weights -> f16, layout preserved [Nout,K] (== B^T, K-contiguous per column)
  wcvt_kernel<<<(1152 * 384 + 255) / 256, 256, 0, stream>>>(qkv_w, wqF, 1152 * 384);
  wcvt_kernel<<<(384 * 384 + 255) / 256, 256, 0, stream>>>(proj_w, wpF, 384 * 384);
  wcvt_kernel<<<(1536 * 384 + 255) / 256, 256, 0, stream>>>(fc1_w, wf1F, 1536 * 384);
  wcvt_kernel<<<(384 * 1536 + 255) / 256, 256, 0, stream>>>(fc2_w, wf2F, 384 * 1536);

  // 2) LN1 + shift + window partition
  ln1_shift_kernel<<<MTOT / 8, 256, 0, stream>>>(x, norm1_g, norm1_b, wins);

  // 3) QKV GEMM: [200704,384] x [384,1152]
  gemm128x64_kernel<<<dim3(1152 / 64, MTOT / 128), 256, 0, stream>>>(
      wins, wqF, qkv_b, Cc, 1152, 0, qbuf, kbuf, vbuf, nullptr, nullptr);

  // 4) attention, one block per (window, head)
  attn_kernel<<<TOTWIN * HEADS, 256, 0, stream>>>(qbuf, kbuf, vbuf, rel_tab, attn_out);

  // 5) proj GEMM + window-reverse + unshift + residual -> x1
  gemm128x64_kernel<<<dim3(Cc / 64, MTOT / 128), 256, 0, stream>>>(
      attn_out, wpF, proj_b, Cc, Cc, 1, nullptr, nullptr, nullptr, x1, x);

  // 6) LN2
  ln2_kernel<<<MTOT / 8, 256, 0, stream>>>(x1, norm2_g, norm2_b, h2);

  // 7) FC1 GEMM + gelu
  gemm128x64_kernel<<<dim3(HIDDEN_ / 64, MTOT / 128), 256, 0, stream>>>(
      h2, wf1F, fc1_b, Cc, HIDDEN_, 2, fc1o, nullptr, nullptr, nullptr, nullptr);

  // 8) FC2 GEMM + residual -> out
  gemm128x64_kernel<<<dim3(Cc / 64, MTOT / 128), 256, 0, stream>>>(
      fc1o, wf2F, fc2_b, HIDDEN_, Cc, 3, nullptr, nullptr, nullptr, out, x1);
}
